// Seq2Seq_26929444946009
// MI455X (gfx1250) — compile-verified, matrix operand order checked by reading, not run on hardware
//
#include <hip/hip_runtime.h>
#include <hip/hip_bf16.h>

// ---------------------------------------------------------------------------
// Seq2Seq GRU encoder/decoder + dot attention + big vocab projection (MI455X)
// Strategy:
//   * Batch all input-side GRU gates (x@Wih^T) up front (no recurrence).
//   * Sequential h-update kernels (one per timestep) using L2-resident Whh.
//   * Attention kernel per decoder step writes combined=[h,ctx] as bf16 rows.
//   * One huge bf16 WMMA GEMM: (B*(T-1)=2528) x (V=32000) x (2H=1024),
//     f32 accumulation, 64x64 wave tile (4x4 v_wmma accumulators) for
//     32.8 FLOP/L2-byte; epilogue scatters into (B,T,V) logits + bias.
// ---------------------------------------------------------------------------

#define Vv 32000
#define Dd 256
#define Hh 512
#define Bb 32
#define Ss 80
#define Tt 80
#define G3H 1536           // 3*H
#define K2H 1024           // 2*H
#define MROWS 2528         // B*(T-1)
#define MPAD  2560         // padded to a multiple of 64 for the GEMM tiling

typedef __attribute__((ext_vector_type(8)))  __bf16 v8bf_t;
typedef __attribute__((ext_vector_type(16))) __bf16 v16bf_t;
typedef __attribute__((ext_vector_type(8)))  float  v8f_t;

// ------------------------- small utility kernels ---------------------------

__global__ void zero_f32(float* __restrict__ p, size_t n) {
    size_t i = (size_t)blockIdx.x * blockDim.x + threadIdx.x;
    size_t st = (size_t)gridDim.x * blockDim.x;
    for (; i < n; i += st) p[i] = 0.0f;
}

// logits[:, 0, :] = 0
__global__ void zero_t0(float* __restrict__ out) {
    size_t i = (size_t)blockIdx.x * blockDim.x + threadIdx.x;
    if (i >= (size_t)Bb * Vv) return;
    size_t b = i / Vv, v = i % Vv;
    out[(b * Tt) * (size_t)Vv + v] = 0.0f;
}

__global__ void cvt_bf16(const float* __restrict__ in, __bf16* __restrict__ out, size_t n) {
    size_t i = (size_t)blockIdx.x * blockDim.x + threadIdx.x;
    size_t st = (size_t)gridDim.x * blockDim.x;
    for (; i < n; i += st) out[i] = (__bf16)in[i];
}

// ----------------- batched input gates: GI = E[tok] @ Wih^T + bih ----------
// One workgroup per (b, step). Embedding row staged in LDS; 256 threads cover
// the 1536 gate outputs (dot length 256).
__global__ __launch_bounds__(256) void gi_kernel(
    const int* __restrict__ tok, int tok_ld, int nsteps,
    const float* __restrict__ E, const float* __restrict__ Wih,
    const float* __restrict__ bih, float* __restrict__ GI) {
    __shared__ float emb[Dd];
    int bs = blockIdx.x;               // b*nsteps + s
    int b = bs / nsteps, s = bs % nsteps;
    int t = tok[b * tok_ld + s];
    int tid = threadIdx.x;
    emb[tid] = E[(size_t)t * Dd + tid];
    __syncthreads();
    float* o = GI + (size_t)bs * G3H;
    for (int j = tid; j < G3H; j += 256) {
        const float* wr = Wih + (size_t)j * Dd;
        float acc = bih[j];
#pragma unroll 8
        for (int d = 0; d < Dd; ++d) acc = fmaf(emb[d], wr[d], acc);
        o[j] = acc;
    }
}

// --------------------- sequential GRU hidden update ------------------------
// grid = (8 jblocks, 32 batch); computes gh = h@Whh^T + bhh for 64 columns
// of all 3 gates, then the PyTorch GRU combine. Optionally records enc_out.
__global__ __launch_bounds__(256) void gru_step(
    const float* __restrict__ hin, float* __restrict__ hout,
    const float* __restrict__ GI, int nsteps, int s,
    const float* __restrict__ Whh, const float* __restrict__ bhh,
    float* __restrict__ seq_out, int seq_ld) {
    __shared__ float hs[Hh];
    __shared__ float gd[192];
    int b = blockIdx.y;
    int j0 = blockIdx.x * 64;
    int tid = threadIdx.x;
    hs[tid]       = hin[(size_t)b * Hh + tid];
    hs[tid + 256] = hin[(size_t)b * Hh + tid + 256];
    __syncthreads();
    if (tid < 192) {
        int g = tid >> 6, u = tid & 63;
        int row = g * Hh + j0 + u;                 // row in (3H x H) Whh
        const float* wr = Whh + (size_t)row * Hh;
        float acc = bhh[row];
#pragma unroll 8
        for (int k = 0; k < Hh; ++k) acc = fmaf(hs[k], wr[k], acc);
        gd[tid] = acc;
    }
    __syncthreads();
    if (tid < 64) {
        int j = j0 + tid;
        const float* gi = GI + ((size_t)b * nsteps + s) * G3H;
        float r = 1.0f / (1.0f + __expf(-(gi[j]        + gd[tid])));
        float z = 1.0f / (1.0f + __expf(-(gi[Hh  + j]  + gd[64  + tid])));
        float n = tanhf(gi[2 * Hh + j] + r * gd[128 + tid]);
        float hn = (1.0f - z) * n + z * hs[j];
        hout[(size_t)b * Hh + j] = hn;
        if (seq_out) seq_out[((size_t)b * seq_ld + s) * Hh + j] = hn;
    }
}

// --------------- attention + combined=[h,ctx] (bf16) per step --------------
__global__ __launch_bounds__(256) void attn_kernel(
    const float* __restrict__ h, const float* __restrict__ enc_out,
    const int* __restrict__ src, __bf16* __restrict__ comb, int trow) {
    __shared__ float hs[Hh];
    __shared__ float sc[Ss];
    __shared__ float inv_sum;
    int b = blockIdx.x, tid = threadIdx.x;
    hs[tid]       = h[(size_t)b * Hh + tid];
    hs[tid + 256] = h[(size_t)b * Hh + tid + 256];
    __syncthreads();
    if (tid < Ss) {
        const float* er = enc_out + ((size_t)b * Ss + tid) * Hh;
        float acc = 0.0f;
#pragma unroll 4
        for (int k = 0; k < Hh; ++k) acc = fmaf(er[k], hs[k], acc);
        sc[tid] = (src[b * Ss + tid] != 0) ? acc : -1e9f;
    }
    __syncthreads();
    if (tid == 0) {
        float m = sc[0];
        for (int sI = 1; sI < Ss; ++sI) m = fmaxf(m, sc[sI]);
        float sum = 0.0f;
        for (int sI = 0; sI < Ss; ++sI) { float e = __expf(sc[sI] - m); sc[sI] = e; sum += e; }
        inv_sum = 1.0f / sum;
    }
    __syncthreads();
    float inv = inv_sum;
    __bf16* crow = comb + ((size_t)trow * Bb + b) * K2H;
    for (int k = tid; k < Hh; k += 256) {
        float ctx = 0.0f;
        for (int sI = 0; sI < Ss; ++sI)
            ctx = fmaf(sc[sI], enc_out[((size_t)b * Ss + sI) * Hh + k], ctx);
        crow[k]      = (__bf16)hs[k];
        crow[Hh + k] = (__bf16)(ctx * inv);
    }
}

// ----------------------- big WMMA projection GEMM --------------------------
// C(2528 x 32000) = A(2528 x 1024, bf16) * W^T  (W row-major NxK bf16) + bias
// f32 accumulate via v_wmma_f32_16x16x32_bf16.
// Per-wave tile: 64(M) x 64(N) = 4x4 accumulators (128 acc VGPRs).
// K loop: 32 steps of 32; 8 fragment loads feed 16 WMMAs per step.

__device__ inline v16bf_t load_frag(const __bf16* p) {
    union { v16bf_t v; v8bf_t h[2]; } u;
    u.h[0] = *(const v8bf_t*)(p);        // halves e0..e7
    u.h[1] = *(const v8bf_t*)(p + 16);   // halves e8..e15
    return u.v;
}

__global__ __launch_bounds__(256) void gemm_out(
    const __bf16* __restrict__ A, const __bf16* __restrict__ W,
    const float* __restrict__ bias, float* __restrict__ out) {
    const int NBLK = Vv / 64;                    // 500
    const int MBLK = MPAD / 64;                  // 40
    int w = blockIdx.x * 8 + (threadIdx.x >> 5); // global wave id (uniform/wave)
    if (w >= MBLK * NBLK) return;
    int lane = threadIdx.x & 31;
    int mblk = w / NBLK, nblk = w % NBLK;
    int m0 = mblk * 64, n0 = nblk * 64;
    int klane = (lane >> 4) << 3;                // 0 or 8 (16-bit A/B layout)
    int mrow  = m0 + (lane & 15);
    int ncol  = lane & 15;

    v8f_t c[4][4] = {};
    const __bf16* ap[4];
    const __bf16* bp[4];
#pragma unroll
    for (int i = 0; i < 4; ++i)
        ap[i] = A + (size_t)(mrow + i * 16) * K2H + klane;
#pragma unroll
    for (int j = 0; j < 4; ++j)
        bp[j] = W + (size_t)(n0 + j * 16 + ncol) * K2H + klane;

    for (int k0 = 0; k0 < K2H; k0 += 32) {
        v16bf_t bf[4];
#pragma unroll
        for (int j = 0; j < 4; ++j) bf[j] = load_frag(bp[j] + k0);
#pragma unroll
        for (int i = 0; i < 4; ++i) {
            v16bf_t a = load_frag(ap[i] + k0);
#pragma unroll
            for (int j = 0; j < 4; ++j)
                c[i][j] = __builtin_amdgcn_wmma_f32_16x16x32_bf16(
                    false, a, false, bf[j], (short)0, c[i][j], false, false);
        }
    }

    // epilogue: row = (t-1)*32 + b  ->  logits[b, t, n]; skip pad rows
#pragma unroll
    for (int i = 0; i < 4; ++i) {
#pragma unroll
        for (int j = 0; j < 4; ++j) {
            int n = n0 + j * 16 + ncol;
            float bb = bias[n];
            int mbase = m0 + i * 16 + ((lane >> 4) << 3);
#pragma unroll
            for (int r = 0; r < 8; ++r) {
                int row = mbase + r;
                if (row < MROWS) {
                    int b = row & 31;
                    int t = (row >> 5) + 1;
                    out[((size_t)b * Tt + t) * Vv + n] = c[i][j][r] + bb;
                }
            }
        }
    }
}

// ---------------------------------------------------------------------------

extern "C" void kernel_launch(void* const* d_in, const int* in_sizes, int n_in,
                              void* d_out, int out_size, void* d_ws, size_t ws_size,
                              hipStream_t stream) {
    const int*   src      = (const int*)  d_in[0];
    const int*   tgt      = (const int*)  d_in[1];
    const float* E        = (const float*)d_in[2];
    const float* enc_Wih  = (const float*)d_in[3];
    const float* enc_Whh  = (const float*)d_in[4];
    const float* enc_bih  = (const float*)d_in[5];
    const float* enc_bhh  = (const float*)d_in[6];
    const float* dec_Wih  = (const float*)d_in[7];
    const float* dec_Whh  = (const float*)d_in[8];
    const float* dec_bih  = (const float*)d_in[9];
    const float* dec_bhh  = (const float*)d_in[10];
    const float* out_W    = (const float*)d_in[11];
    const float* out_b    = (const float*)d_in[12];
    float* out = (float*)d_out;

    // workspace carve-up (256B aligned)
    char* ws = (char*)d_ws;
    size_t off = 0;
    auto take = [&](size_t bytes) -> char* {
        char* p = ws + off;
        off = (off + bytes + 255) & ~(size_t)255;
        return p;
    };
    float*  GI_enc  = (float*) take((size_t)Bb * Ss * G3H * 4);        // 15.7 MB
    float*  GI_dec  = (float*) take((size_t)Bb * (Tt - 1) * G3H * 4);  // 15.5 MB
    float*  enc_out = (float*) take((size_t)Bb * Ss * Hh * 4);         //  5.2 MB
    float*  h0      = (float*) take((size_t)Bb * Hh * 4);
    float*  h1      = (float*) take((size_t)Bb * Hh * 4);
    __bf16* comb    = (__bf16*)take((size_t)MPAD * K2H * 2);           //  5.2 MB
    __bf16* Wbf     = (__bf16*)take((size_t)Vv * K2H * 2);             // 65.5 MB
    (void)ws_size;

    // 1. one-time conversions / zeroing
    cvt_bf16<<<8192, 256, 0, stream>>>(out_W, Wbf, (size_t)Vv * K2H);
    zero_f32<<<64, 256, 0, stream>>>(h0, (size_t)Bb * Hh);
    zero_t0<<<(Bb * Vv + 255) / 256, 256, 0, stream>>>(out);

    // 2. batched input-side gates (no recurrence)
    gi_kernel<<<Bb * Ss, 256, 0, stream>>>(src, Ss, Ss, E, enc_Wih, enc_bih, GI_enc);
    gi_kernel<<<Bb * (Tt - 1), 256, 0, stream>>>(tgt, Tt, Tt - 1, E, dec_Wih, dec_bih, GI_dec);

    // 3. encoder recurrence (80 sequential steps), records enc_out
    float* hA = h0;
    float* hB = h1;
    for (int s = 0; s < Ss; ++s) {
        gru_step<<<dim3(Hh / 64, Bb), 256, 0, stream>>>(
            hA, hB, GI_enc, Ss, s, enc_Whh, enc_bhh, enc_out, Ss);
        float* tmp = hA; hA = hB; hB = tmp;
    }
    // final encoder hidden now in hA

    // 4. decoder recurrence + attention; emit combined=[h,ctx] bf16 rows
    for (int t = 1; t < Tt; ++t) {
        gru_step<<<dim3(Hh / 64, Bb), 256, 0, stream>>>(
            hA, hB, GI_dec, Tt - 1, t - 1, dec_Whh, dec_bhh, nullptr, 0);
        attn_kernel<<<Bb, 256, 0, stream>>>(hB, enc_out, src, comb, t - 1);
        float* tmp = hA; hA = hB; hB = tmp;
    }

    // 5. one big WMMA GEMM: logits[:,1:,:] = combined @ out_W^T + out_b
    const int njobs = (MPAD / 64) * (Vv / 64);           // 20000 wave tiles
    gemm_out<<<(njobs + 7) / 8, 256, 0, stream>>>(comb, Wbf, out_b, out);
}
